// Attention_LoRA_88725434401372
// MI455X (gfx1250) — matrix operand order repaired and checked by reading, not compiled, not run on hardware
//
#include <hip/hip_runtime.h>
#include <stdint.h>

#define B_ 2
#define S_ 2048
#define P_ 1024
#define D_ 2048
#define H_ 16
#define R_ 16
#define DH_ 128
#define KV_ (P_ + S_)   /* 3072 */
#define MTOT (B_ * S_)  /* 4096 */

typedef __attribute__((ext_vector_type(16))) __bf16 v16bf;
typedef __attribute__((ext_vector_type(8)))  float  v8f;

union Frag { v16bf v; unsigned int u[8]; };

// Packed f32 -> bf16 conversion on the CDNA5 VALU (RNE), 1 instruction.
__device__ __forceinline__ unsigned int pk2(float a, float b) {
  unsigned int r;
  asm("v_cvt_pk_bf16_f32 %0, %1, %2" : "=v"(r) : "v"(a), "v"(b));
  return r;
}
__device__ __forceinline__ unsigned short f2bf(float f) {
  return (unsigned short)(pk2(f, 0.f) & 0xFFFFu);
}
__device__ __forceinline__ float bf2f(unsigned short h) {
  return __uint_as_float(((unsigned int)h) << 16);
}
// A-matrix 16x32 bf16 K index for VGPR j, lane-half kh (ISA 7.12.2)
__device__ __forceinline__ int kpat(int j, int kh) {
  return ((j >= 4) ? 16 : 0) + (j & 3) * 2 + kh * 8;
}
__device__ __forceinline__ v8f vzero() {
  v8f r;
#pragma unroll
  for (int j = 0; j < 8; ++j) r[j] = 0.f;
  return r;
}
__device__ __forceinline__ v8f wmma_bf16(const Frag& a, const Frag& b, v8f c) {
  return __builtin_amdgcn_wmma_f32_16x16x32_bf16(false, a.v, false, b.v,
                                                 (short)0, c, false, false);
}

// CDNA5 async global->LDS copy (ASYNCcnt-tracked), 16B per lane.
__device__ __forceinline__ void async_b128(unsigned lds_off, const void* gaddr) {
  asm volatile("global_load_async_to_lds_b128 %0, %1, off"
               :: "v"(lds_off), "v"(gaddr)
               : "memory");
}
__device__ __forceinline__ void wait_async0() {
  asm volatile("s_wait_asynccnt 0" ::: "memory");
}
__device__ __forceinline__ unsigned lds_off32(const void* p) {
  return (unsigned)(size_t)p;   // flat LDS aperture: addr[31:0] = LDS offset
}

// ---------------------------------------------------------------- prev KV copy
__global__ void copy_prev_kernel(const float* __restrict__ pk,
                                 const float* __restrict__ pv,
                                 unsigned short* __restrict__ kb,
                                 unsigned short* __restrict__ vb) {
  int idx = blockIdx.x * blockDim.x + threadIdx.x;   // over B*P*H*DH = 2^22
  int d = idx & (DH_ - 1);
  int h = (idx >> 7) & (H_ - 1);
  int p = (idx >> 11) & (P_ - 1);
  int b = idx >> 21;
  int dst = ((b * H_ + h) * KV_ + p) * DH_ + d;
  kb[dst] = f2bf(pk[idx]);
  vb[dst] = f2bf(pv[idx]);
}

// ------------------------------------------------------------- LoRA down (x3)
__global__ void lora_down3_kernel(const float* __restrict__ x,
                                  const float* __restrict__ A0,
                                  const float* __restrict__ A1,
                                  const float* __restrict__ A2,
                                  float* __restrict__ o0,
                                  float* __restrict__ o1,
                                  float* __restrict__ o2) {
  int m = blockIdx.x;
  int tid = threadIdx.x, lane = tid & 31, wid = tid >> 5;
  float acc[48];
#pragma unroll
  for (int i = 0; i < 48; ++i) acc[i] = 0.f;
  for (int k = tid; k < D_; k += 128) {
    float xv = x[m * D_ + k];
#pragma unroll
    for (int r = 0; r < 16; ++r) {
      acc[r]      += xv * A0[k * 16 + r];
      acc[16 + r] += xv * A1[k * 16 + r];
      acc[32 + r] += xv * A2[k * 16 + r];
    }
  }
  __shared__ float red[4][48];
#pragma unroll
  for (int i = 0; i < 48; ++i) {
    float a = acc[i];
#pragma unroll
    for (int off = 16; off >= 1; off >>= 1) a += __shfl_xor(a, off, 32);
    if (lane == 0) red[wid][i] = a;
  }
  __syncthreads();
  if (tid < 48) {
    float s = red[0][tid] + red[1][tid] + red[2][tid] + red[3][tid];
    int mat = tid >> 4, r = tid & 15;
    float* o = (mat == 0) ? o0 : (mat == 1) ? o1 : o2;
    o[m * 16 + r] = s;
  }
}

// -------------------------------------------------------- LoRA down (bf16 in)
__global__ void lora_down1_kernel(const unsigned short* __restrict__ x,
                                  const float* __restrict__ A,
                                  float* __restrict__ o) {
  int m = blockIdx.x;
  int tid = threadIdx.x, lane = tid & 31, wid = tid >> 5;
  float acc[16];
#pragma unroll
  for (int i = 0; i < 16; ++i) acc[i] = 0.f;
  for (int k = tid; k < D_; k += 128) {
    float xv = bf2f(x[m * D_ + k]);
#pragma unroll
    for (int r = 0; r < 16; ++r) acc[r] += xv * A[k * 16 + r];
  }
  __shared__ float red[4][16];
#pragma unroll
  for (int i = 0; i < 16; ++i) {
    float a = acc[i];
#pragma unroll
    for (int off = 16; off >= 1; off >>= 1) a += __shfl_xor(a, off, 32);
    if (lane == 0) red[wid][i] = a;
  }
  __syncthreads();
  if (tid < 16) o[m * 16 + tid] = red[0][tid] + red[1][tid] + red[2][tid] + red[3][tid];
}

// ------------------------------------- fused QKV projection + LoRA + RoPE + scatter
__global__ __launch_bounds__(256) void proj_kernel(
    const float* __restrict__ x,
    const float* __restrict__ wq, const float* __restrict__ wk, const float* __restrict__ wv,
    const float* __restrict__ xaq, const float* __restrict__ xak, const float* __restrict__ xav,
    const float* __restrict__ bq, const float* __restrict__ bk, const float* __restrict__ bv,
    const float* __restrict__ fc, const float* __restrict__ fs,
    const float* __restrict__ ps,
    unsigned short* __restrict__ qb, unsigned short* __restrict__ kb,
    unsigned short* __restrict__ vb) {
  const int z = blockIdx.z;
  const float* W  = (z == 0) ? wq  : (z == 1) ? wk  : wv;
  const float* XA = (z == 0) ? xaq : (z == 1) ? xak : xav;
  const float* WB = (z == 0) ? bq  : (z == 1) ? bk  : bv;
  const int m0 = blockIdx.y * 128, n0 = blockIdx.x * 64;
  const int tid = threadIdx.x, lane = tid & 31, wid = tid >> 5;
  const int wm = wid >> 1, wn = wid & 1, ln15 = lane & 15, kh = lane >> 4;

  __shared__ __align__(16) unsigned char smem[12288];
  unsigned short* As = (unsigned short*)smem;            // 128x32 bf16
  unsigned short* Ws = (unsigned short*)(smem + 8192);   // 64x32 bf16

  v8f acc[2][2];
#pragma unroll
  for (int i = 0; i < 2; ++i)
#pragma unroll
    for (int j = 0; j < 2; ++j) acc[i][j] = vzero();

  for (int kk = 0; kk < D_; kk += 32) {
    __syncthreads();
    {
      int row = tid >> 1, seg = tid & 1;
      const float4* src = (const float4*)&x[(m0 + row) * D_ + kk + seg * 16];
      float4 a = src[0], b = src[1], c = src[2], d = src[3];
      uint4* dst = (uint4*)&As[row * 32 + seg * 16];
      dst[0] = make_uint4(pk2(a.x, a.y), pk2(a.z, a.w), pk2(b.x, b.y), pk2(b.z, b.w));
      dst[1] = make_uint4(pk2(c.x, c.y), pk2(c.z, c.w), pk2(d.x, d.y), pk2(d.z, d.w));
      int wrow = tid >> 2, wseg = tid & 3;
      const float4* wsrc = (const float4*)&W[(n0 + wrow) * D_ + kk + wseg * 8];
      float4 e = wsrc[0], f = wsrc[1];
      *(uint4*)&Ws[wrow * 32 + wseg * 8] =
          make_uint4(pk2(e.x, e.y), pk2(e.z, e.w), pk2(f.x, f.y), pk2(f.z, f.w));
      if (kk + 32 < D_) {   // prefetch next K-step (GL2 stream)
        __builtin_prefetch(&x[(m0 + row) * D_ + kk + 32 + seg * 16], 0, 1);
        __builtin_prefetch(&W[(n0 + wrow) * D_ + kk + 32 + wseg * 8], 0, 1);
      }
    }
    __syncthreads();
    const unsigned int* As32 = (const unsigned int*)As;
    const unsigned int* Ws32 = (const unsigned int*)Ws;
    Frag a[2], b[2];
#pragma unroll
    for (int mi = 0; mi < 2; ++mi) {
      int row = wm * 32 + mi * 16 + ln15;
#pragma unroll
      for (int j = 0; j < 8; ++j) a[mi].u[j] = As32[(row * 32 + kpat(j, kh)) >> 1];
    }
#pragma unroll
    for (int ni = 0; ni < 2; ++ni) {
      int nrow = wn * 32 + ni * 16 + ln15;
#pragma unroll
      for (int j = 0; j < 8; ++j) b[ni].u[j] = Ws32[(nrow * 32 + kpat(j, kh)) >> 1];
    }
#pragma unroll
    for (int mi = 0; mi < 2; ++mi)
#pragma unroll
      for (int ni = 0; ni < 2; ++ni) acc[mi][ni] = wmma_bf16(a[mi], b[ni], acc[mi][ni]);
  }

  // epilogue: LoRA tiles into (reused) LDS
  __syncthreads();
  float* xs  = (float*)smem;            // 128 x 16
  float* wbs = (float*)(smem + 8192);   // 16 x 64
  {
    int row = tid >> 1, hseg = tid & 1;
    const float4* src = (const float4*)&XA[(m0 + row) * 16 + hseg * 8];
    *(float4*)&xs[row * 16 + hseg * 8]     = src[0];
    *(float4*)&xs[row * 16 + hseg * 8 + 4] = src[1];
    int idx = tid * 4;                  // 16*64 floats over 256 threads
    int r = idx >> 6, c = idx & 63;
    *(float4*)&wbs[r * 64 + c] = *(const float4*)&WB[r * D_ + n0 + c];
  }
  __syncthreads();
  const int bidx = m0 >> 11;
  const float psv = ps[bidx];
  const bool rope = (z != 2);
#pragma unroll
  for (int mi = 0; mi < 2; ++mi)
#pragma unroll
    for (int ni = 0; ni < 2; ++ni)
#pragma unroll
      for (int j = 0; j < 8; ++j) {
        int ml = wm * 32 + mi * 16 + j + (kh ? 8 : 0);
        int nl = wn * 32 + ni * 16 + ln15;
        float lr = 0.f;
#pragma unroll
        for (int r = 0; r < 16; ++r) lr += xs[ml * 16 + r] * wbs[r * 64 + nl];
        float val = acc[mi][ni][j] + psv * lr;
        int s = (m0 + ml) & (S_ - 1);
        int n = n0 + nl;
        int h = n >> 7, d = n & 127;
        if (rope) {
          int p = d >> 1;
          float co = fc[s * 64 + p], sn = fs[s * 64 + p];
          float part = __shfl_xor(val, 1, 32);
          val = (d & 1) ? (part * sn + val * co) : (val * co - part * sn);
        }
        unsigned short ob = f2bf(val);
        if (z == 0)      qb[((bidx * H_ + h) * S_ + s) * DH_ + d] = ob;
        else if (z == 1) kb[((bidx * H_ + h) * KV_ + P_ + s) * DH_ + d] = ob;
        else             vb[((bidx * H_ + h) * KV_ + P_ + s) * DH_ + d] = ob;
      }
}

// ------------------------------------------------------------ flash attention
__global__ __launch_bounds__(128) void attn_kernel(
    const unsigned short* __restrict__ qb, const unsigned short* __restrict__ kbuf,
    const unsigned short* __restrict__ vbuf, unsigned short* __restrict__ ctx) {
  const int tid = threadIdx.x, lane = tid & 31, wid = tid >> 5;
  const int ln15 = lane & 15, kh = lane >> 4;
  const int h = blockIdx.y, bz = blockIdx.z;
  const int q0 = blockIdx.x * 64;
  const int q0w = q0 + wid * 16;

  __shared__ unsigned short Ks[32 * 128];   // [key][dh]
  __shared__ unsigned short Vt[128 * 32];   // [dh][key]
  __shared__ unsigned short Ps[4][16 * 32]; // per-wave probs [m][key]

  Frag qf[4];
  {
    const unsigned int* qg = (const unsigned int*)qb;
    int base = ((bz * H_ + h) * S_ + q0w + ln15) * DH_;
#pragma unroll
    for (int c = 0; c < 4; ++c)
#pragma unroll
      for (int j = 0; j < 8; ++j) qf[c].u[j] = qg[(base + c * 32 + kpat(j, kh)) >> 1];
  }
  float mx[8], ls[8];
  v8f o[8];
#pragma unroll
  for (int j = 0; j < 8; ++j) { mx[j] = -1e30f; ls[j] = 0.f; }
#pragma unroll
  for (int t8 = 0; t8 < 8; ++t8) o[t8] = vzero();

  const int nkb = (P_ + q0 + 64) / 32;
  const unsigned short* kg = kbuf + (size_t)((bz * H_ + h) * KV_) * DH_;
  const unsigned short* vg = vbuf + (size_t)((bz * H_ + h) * KV_) * DH_;

  for (int kb2 = 0; kb2 < nkb; ++kb2) {
    __syncthreads();
    {
      int krow = tid >> 2, seg = tid & 3;
      int goff = (kb2 * 32 + krow) * DH_ + seg * 32;
      // K tile: raw bf16 bytes -> LDS via CDNA5 async copy (no VGPR round-trip)
#pragma unroll
      for (int c = 0; c < 4; ++c)
        async_b128(lds_off32(&Ks[krow * 128 + seg * 32 + c * 8]), &kg[goff + c * 8]);
      // V tile: needs dh-major transpose, so load to VGPRs and scatter
      const uint4* srcv = (const uint4*)&vg[goff];
      union { uint4 q[4]; unsigned short s[32]; } t;
      t.q[0] = srcv[0]; t.q[1] = srcv[1]; t.q[2] = srcv[2]; t.q[3] = srcv[3];
#pragma unroll
      for (int i = 0; i < 32; ++i) Vt[(seg * 32 + i) * 32 + krow] = t.s[i];
    }
    wait_async0();
    __syncthreads();
    const unsigned int* Ks32 = (const unsigned int*)Ks;
    const unsigned int* Vt32 = (const unsigned int*)Vt;

    v8f sc[2];
    sc[0] = vzero(); sc[1] = vzero();
#pragma unroll
    for (int ni = 0; ni < 2; ++ni) {
      int keyrow = ni * 16 + ln15;
#pragma unroll
      for (int c = 0; c < 4; ++c) {
        Frag bfr;
#pragma unroll
        for (int j = 0; j < 8; ++j)
          bfr.u[j] = Ks32[(keyrow * 128 + c * 32 + kpat(j, kh)) >> 1];
        sc[ni] = wmma_bf16(qf[c], bfr, sc[ni]);
      }
    }
    // online softmax
#pragma unroll
    for (int j = 0; j < 8; ++j) {
      int mrow = j + (kh ? 8 : 0);
      int qi = q0w + mrow;
      int kj0 = kb2 * 32 + ln15;
      float s0 = sc[0][j] * 0.08838834764831845f + ((kj0 > P_ + qi) ? -1e9f : 0.f);
      float s1 = sc[1][j] * 0.08838834764831845f + ((kj0 + 16 > P_ + qi) ? -1e9f : 0.f);
      float t = fmaxf(s0, s1);
      t = fmaxf(t, __shfl_xor(t, 1, 32));
      t = fmaxf(t, __shfl_xor(t, 2, 32));
      t = fmaxf(t, __shfl_xor(t, 4, 32));
      t = fmaxf(t, __shfl_xor(t, 8, 32));
      float mn = fmaxf(mx[j], t);
      float sf = __expf(mx[j] - mn);
      mx[j] = mn;
      float e0 = __expf(s0 - mn), e1 = __expf(s1 - mn);
      float rs = e0 + e1;
      rs += __shfl_xor(rs, 1, 32);
      rs += __shfl_xor(rs, 2, 32);
      rs += __shfl_xor(rs, 4, 32);
      rs += __shfl_xor(rs, 8, 32);
      ls[j] = ls[j] * sf + rs;
#pragma unroll
      for (int t8 = 0; t8 < 8; ++t8) o[t8][j] *= sf;
      unsigned int ppk = pk2(e0, e1);   // one packed bf16 convert for both probs
      Ps[wid][mrow * 32 + ln15] = (unsigned short)(ppk & 0xFFFFu);
      Ps[wid][mrow * 32 + 16 + ln15] = (unsigned short)(ppk >> 16);
    }
    Frag pa;
    const unsigned int* Ps32 = (const unsigned int*)Ps[wid];
#pragma unroll
    for (int j = 0; j < 8; ++j) pa.u[j] = Ps32[(ln15 * 32 + kpat(j, kh)) >> 1];
#pragma unroll
    for (int t8 = 0; t8 < 8; ++t8) {
      Frag bfr;
      int dh = t8 * 16 + ln15;
#pragma unroll
      for (int j = 0; j < 8; ++j) bfr.u[j] = Vt32[(dh * 32 + kpat(j, kh)) >> 1];
      o[t8] = wmma_bf16(pa, bfr, o[t8]);
    }
  }
#pragma unroll
  for (int t8 = 0; t8 < 8; ++t8)
#pragma unroll
    for (int j = 0; j < 8; ++j) {
      int mrow = j + (kh ? 8 : 0);
      int s = q0w + mrow;
      int d = h * DH_ + t8 * 16 + ln15;
      ctx[(size_t)(bz * S_ + s) * D_ + d] = f2bf(o[t8][j] / ls[j]);
    }
}

// ---------------------------------------------------- output projection + LoRA
__global__ __launch_bounds__(256) void outproj_kernel(
    const unsigned short* __restrict__ ctx, const float* __restrict__ wo,
    const float* __restrict__ xao, const float* __restrict__ wob,
    const float* __restrict__ ps, float* __restrict__ out) {
  const int m0 = blockIdx.y * 128, n0 = blockIdx.x * 64;
  const int tid = threadIdx.x, lane = tid & 31, wid = tid >> 5;
  const int wm = wid >> 1, wn = wid & 1, ln15 = lane & 15, kh = lane >> 4;

  __shared__ __align__(16) unsigned char smem[12288];
  unsigned short* As = (unsigned short*)smem;
  unsigned short* Ws = (unsigned short*)(smem + 8192);

  v8f acc[2][2];
#pragma unroll
  for (int i = 0; i < 2; ++i)
#pragma unroll
    for (int j = 0; j < 2; ++j) acc[i][j] = vzero();

  for (int kk = 0; kk < D_; kk += 32) {
    __syncthreads();
    {
      int row = tid >> 1, seg = tid & 1;
      // ctx tile is already bf16: pure byte copy -> CDNA5 async global->LDS
      const unsigned short* srcp = &ctx[(size_t)(m0 + row) * D_ + kk + seg * 16];
      async_b128(lds_off32(&As[row * 32 + seg * 16]), srcp);
      async_b128(lds_off32(&As[row * 32 + seg * 16 + 8]), srcp + 8);
      int wrow = tid >> 2, wseg = tid & 3;
      const float4* wsrc = (const float4*)&wo[(n0 + wrow) * D_ + kk + wseg * 8];
      float4 e = wsrc[0], f = wsrc[1];
      *(uint4*)&Ws[wrow * 32 + wseg * 8] =
          make_uint4(pk2(e.x, e.y), pk2(e.z, e.w), pk2(f.x, f.y), pk2(f.z, f.w));
      if (kk + 32 < D_) {
        __builtin_prefetch(srcp + 32, 0, 1);   // next K-chunk of this ctx row
        __builtin_prefetch(&wo[(n0 + wrow) * D_ + kk + 32 + wseg * 8], 0, 1);
      }
    }
    wait_async0();
    __syncthreads();
    const unsigned int* As32 = (const unsigned int*)As;
    const unsigned int* Ws32 = (const unsigned int*)Ws;
    Frag a[2], b[2];
#pragma unroll
    for (int mi = 0; mi < 2; ++mi) {
      int row = wm * 32 + mi * 16 + ln15;
#pragma unroll
      for (int j = 0; j < 8; ++j) a[mi].u[j] = As32[(row * 32 + kpat(j, kh)) >> 1];
    }
#pragma unroll
    for (int ni = 0; ni < 2; ++ni) {
      int nrow = wn * 32 + ni * 16 + ln15;
#pragma unroll
      for (int j = 0; j < 8; ++j) b[ni].u[j] = Ws32[(nrow * 32 + kpat(j, kh)) >> 1];
    }
#pragma unroll
    for (int mi = 0; mi < 2; ++mi)
#pragma unroll
      for (int ni = 0; ni < 2; ++ni) acc[mi][ni] = wmma_bf16(a[mi], b[ni], acc[mi][ni]);
  }

  __syncthreads();
  float* xs  = (float*)smem;
  float* wbs = (float*)(smem + 8192);
  {
    int row = tid >> 1, hseg = tid & 1;
    const float4* src = (const float4*)&xao[(m0 + row) * 16 + hseg * 8];
    *(float4*)&xs[row * 16 + hseg * 8]     = src[0];
    *(float4*)&xs[row * 16 + hseg * 8 + 4] = src[1];
    int idx = tid * 4;
    int r = idx >> 6, c = idx & 63;
    *(float4*)&wbs[r * 64 + c] = *(const float4*)&wob[r * D_ + n0 + c];
  }
  __syncthreads();
  const float psv = ps[m0 >> 11];
#pragma unroll
  for (int mi = 0; mi < 2; ++mi)
#pragma unroll
    for (int ni = 0; ni < 2; ++ni)
#pragma unroll
      for (int j = 0; j < 8; ++j) {
        int ml = wm * 32 + mi * 16 + j + (kh ? 8 : 0);
        int nl = wn * 32 + ni * 16 + ln15;
        float lr = 0.f;
#pragma unroll
        for (int r = 0; r < 16; ++r) lr += xs[ml * 16 + r] * wbs[r * 64 + nl];
        out[(size_t)(m0 + ml) * D_ + n0 + nl] = acc[mi][ni][j] + psv * lr;
      }
}

// ------------------------------------------------------------------- launcher
extern "C" void kernel_launch(void* const* d_in, const int* in_sizes, int n_in,
                              void* d_out, int out_size, void* d_ws, size_t ws_size,
                              hipStream_t stream) {
  (void)in_sizes; (void)n_in; (void)out_size; (void)ws_size;
  const float* x   = (const float*)d_in[0];
  const float* fc  = (const float*)d_in[1];
  const float* fs  = (const float*)d_in[2];
  const float* pk  = (const float*)d_in[4];
  const float* pv  = (const float*)d_in[5];
  const float* ps  = (const float*)d_in[6];
  const float* wq  = (const float*)d_in[7];
  const float* wk  = (const float*)d_in[8];
  const float* wv  = (const float*)d_in[9];
  const float* wo  = (const float*)d_in[10];
  const float* wqA = (const float*)d_in[11];
  const float* wqB = (const float*)d_in[12];
  const float* wkA = (const float*)d_in[13];
  const float* wkB = (const float*)d_in[14];
  const float* wvA = (const float*)d_in[15];
  const float* wvB = (const float*)d_in[16];
  const float* woA = (const float*)d_in[17];
  const float* woB = (const float*)d_in[18];

  unsigned char* w = (unsigned char*)d_ws;
  unsigned short* qb  = (unsigned short*)(w);                 // 16 MiB
  unsigned short* kb  = (unsigned short*)(w + 16777216);      // 24 MiB
  unsigned short* vb  = (unsigned short*)(w + 41943040);      // 24 MiB
  unsigned short* ctx = (unsigned short*)(w + 67108864);      // 16 MiB
  float* xaq = (float*)(w + 83886080);
  float* xak = (float*)(w + 84148224);
  float* xav = (float*)(w + 84410368);
  float* xao = (float*)(w + 84672512);                        // end ~81 MiB

  copy_prev_kernel<<<(B_ * P_ * H_ * DH_) / 256, 256, 0, stream>>>(pk, pv, kb, vb);
  lora_down3_kernel<<<MTOT, 128, 0, stream>>>(x, wqA, wkA, wvA, xaq, xak, xav);
  proj_kernel<<<dim3(D_ / 64, MTOT / 128, 3), 256, 0, stream>>>(
      x, wq, wk, wv, xaq, xak, xav, wqB, wkB, wvB, fc, fs, ps, qb, kb, vb);
  attn_kernel<<<dim3(S_ / 64, H_, B_), 128, 0, stream>>>(qb, kb, vb, ctx);
  lora_down1_kernel<<<MTOT, 128, 0, stream>>>(ctx, woA, xao);
  outproj_kernel<<<dim3(D_ / 64, MTOT / 128, 1), 256, 0, stream>>>(
      ctx, wo, xao, woB, ps, (float*)d_out);
}
